// MultiHeadAttention_43164421325430
// MI455X (gfx1250) — matrix operand order repaired and checked by reading, not compiled, not run on hardware
//
#include <hip/hip_runtime.h>
#include <hip/hip_bf16.h>

#define BB 4
#define SS 2048
#define DD 1024
#define HH 16
#define HDIM 64
#define MTOT (BB * SS)

// Tensor Data Mover staging (inline asm -> portable across toolchains).
#define USE_TDM 1

typedef __attribute__((ext_vector_type(16))) __bf16 v16bf;
typedef __attribute__((ext_vector_type(8))) float v8f;
typedef __attribute__((ext_vector_type(4))) unsigned int v4u;
typedef __attribute__((ext_vector_type(8))) int v8i;
typedef __attribute__((ext_vector_type(4))) int v4i;

union FragAB {
    v16bf v;
    uint4 q[2];
};
union FragC {
    v8f v;
    float f[8];
};

__device__ __forceinline__ unsigned short f32_bf16(float x) {
    union { float f; unsigned u; } c;
    c.f = x;
    unsigned r = c.u + 0x7fffu + ((c.u >> 16) & 1u);
    return (unsigned short)(r >> 16);
}

#if USE_TDM
// ---------------------------------------------------------------------------
// TDM: DMA a 2D tile of bf16 (data_size=2B) from global to LDS with row pad.
//   tile_w elems per row, tile_h rows, global row stride in elems.
//   pad_interval_code: pad after 2^(code+1) DWORDs; pad_amount_code: (code+1) DW.
// D# layout per CDNA5 ISA ch.8 (group0 128b, group1 256b, groups 2/3 zero = 2D).
// Issued as inline asm (VADDR0=4 SGPRs, VADDR1=8 SGPRs, VADDR2/3=4 SGPRs),
// bypassing the toolchain-dependent clang builtin.
// ---------------------------------------------------------------------------
__device__ __forceinline__ void tdm_load_2d_bf16(
    const void* gsrc, const void* lds_dst, unsigned tile_w, unsigned tile_h,
    unsigned long long row_stride_elems, unsigned tensor_w, unsigned tensor_h,
    unsigned pad_interval_code, unsigned pad_amount_code) {
#if defined(__HIP_DEVICE_COMPILE__)
    unsigned long long ga = (unsigned long long)(size_t)gsrc;
    unsigned lds = (unsigned)(size_t)lds_dst;  // low 32 bits = LDS offset
    v4u g0;
    g0[0] = 1u;                                       // count=1 (valid), user mode
    g0[1] = lds;                                      // [63:32] lds_addr
    g0[2] = (unsigned)ga;                             // [95:64] global_addr lo
    g0[3] = (unsigned)((ga >> 32) & 0x01ffffffull)    // [120:96] global_addr hi
            | (2u << 30);                             // [127:126] type=2 (image)
    v8i g1;
    g1[0] = (int)((1u << 16)                          // data_size=1 -> 2 bytes
                  | (1u << 20)                        // pad_enable
                  | (pad_interval_code << 22)         // [24:22]
                  | (pad_amount_code << 25));         // [31:25]
    g1[1] = (int)((tensor_w & 0xffffu) << 16);        // [63:48] tensor_dim0 lo16
    g1[2] = (int)((tensor_w >> 16) | ((tensor_h & 0xffffu) << 16));
    g1[3] = (int)((tensor_h >> 16) | (tile_w << 16)); // [127:112] tile_dim0
    g1[4] = (int)tile_h;                              // [143:128] tile_dim1, dim2=0
    g1[5] = (int)(unsigned)(row_stride_elems & 0xffffffffull);
    g1[6] = (int)(unsigned)((row_stride_elems >> 32) & 0xffffull);
    g1[7] = 0;
    v4i g2 = {0, 0, 0, 0};
    v4i g3 = {0, 0, 0, 0};
    asm volatile("tensor_load_to_lds %0, %1, %2, %3"
                 :
                 : "s"(g0), "s"(g1), "s"(g2), "s"(g3)
                 : "memory");
#endif
}

__device__ __forceinline__ void tdm_wait() {
#if defined(__HIP_DEVICE_COMPILE__)
#if __has_builtin(__builtin_amdgcn_s_wait_tensorcnt)
    __builtin_amdgcn_s_wait_tensorcnt(0);
#else
    asm volatile("s_wait_tensorcnt 0x0" ::: "memory");
#endif
#endif
}
#endif  // USE_TDM

// ---------------------------------------------------------------------------
// f32 -> bf16 conversion, 8 elements per thread, vectorized 16B store
// ---------------------------------------------------------------------------
__global__ __launch_bounds__(256) void cvt_f32_bf16_kernel(
    const float* __restrict__ in, unsigned short* __restrict__ out, int n8) {
    int i = blockIdx.x * blockDim.x + threadIdx.x;
    if (i >= n8) return;
    const float4* p = (const float4*)in + (size_t)i * 2;
    float4 a = p[0];
    float4 b = p[1];
    union { unsigned short us[8]; uint4 q; } o;
    o.us[0] = f32_bf16(a.x); o.us[1] = f32_bf16(a.y);
    o.us[2] = f32_bf16(a.z); o.us[3] = f32_bf16(a.w);
    o.us[4] = f32_bf16(b.x); o.us[5] = f32_bf16(b.y);
    o.us[6] = f32_bf16(b.z); o.us[7] = f32_bf16(b.w);
    ((uint4*)out)[i] = o.q;
}

// ---------------------------------------------------------------------------
// GEMM: out[m,n] = sum_k A[m,k] * W[n,k] + bias[n]
//   A: [M=8192, K=1024] bf16 row-major, W: [N=1024, K=1024] bf16 row-major.
// Workgroup tile 128x128, 8 waves, wave tile 64x32 (4x2 WMMA 16x16 tiles).
// A staged in double-buffered LDS via TDM: TDM for slab k0+32 is issued
// BEFORE computing slab k0, so the DMA overlaps the WMMAs; the
// s_wait_tensorcnt + barrier come after compute.
// mode 0: f32 out row-major [M,N]      (final projection -> d_out)
// mode 1: bf16 out [B,H,S,HD]          (Q,K head layout)
// mode 2: bf16 out [B,H,HD,S]          (V transposed head layout)
// ---------------------------------------------------------------------------
__global__ __launch_bounds__(256, 1) void gemm_bf16_kernel(
    const unsigned short* __restrict__ A, const unsigned short* __restrict__ W,
    const float* __restrict__ bias, void* __restrict__ out, int mode) {
    const int K = DD, N = DD;
    __shared__ unsigned short sA[2][128 * 40];  // double-buffered 32-wide K slab

    const int tile_m = blockIdx.x * 128;
    const int tile_n = blockIdx.y * 128;
    const int tid = threadIdx.x;
    const int lane = tid & 31;
    const int wave = tid >> 5;
    const int waveM = wave >> 2;  // 0..1
    const int waveN = wave & 3;   // 0..3
    const int l16 = lane & 15;
    const int lhalf = lane >> 4;

    FragC c[4][2];
#pragma unroll
    for (int i = 0; i < 4; i++)
#pragma unroll
        for (int j = 0; j < 2; j++)
#pragma unroll
            for (int r = 0; r < 8; r++) c[i][j].f[r] = 0.f;

#if !USE_TDM
    const int srow = tid >> 1;
    const int sseg = tid & 1;
    const unsigned short* aRow = A + (size_t)(tile_m + srow) * K + sseg * 16;
#endif

    // B fragment base: lane reads W[n][k0 + lhalf*8 ..] and W[n][k0 + 16 + lhalf*8 ..]
    const unsigned short* wBase[2];
#pragma unroll
    for (int sn = 0; sn < 2; sn++)
        wBase[sn] = W + (size_t)(tile_n + waveN * 32 + sn * 16 + l16) * K + lhalf * 8;

    // prologue: stage slab 0 into buffer 0
#if USE_TDM
    tdm_load_2d_bf16(A + (size_t)(tile_m + wave * 16) * K, &sA[0][wave * 16 * 40],
                     32, 16, (unsigned long long)K, K, MTOT, 3, 3);
    tdm_wait();
#else
    *(uint4*)&sA[0][srow * 40 + sseg * 16] = *(const uint4*)(aRow);
    *(uint4*)&sA[0][srow * 40 + sseg * 16 + 8] = *(const uint4*)(aRow + 8);
#endif
    __syncthreads();

    for (int k0 = 0; k0 < K; k0 += 32) {
        const int p = (k0 >> 5) & 1;
        // issue next slab's DMA into the other buffer (overlaps compute below)
        if (k0 + 32 < K) {
#if USE_TDM
            tdm_load_2d_bf16(A + (size_t)(tile_m + wave * 16) * K + k0 + 32,
                             &sA[1 - p][wave * 16 * 40],
                             32, 16, (unsigned long long)K, K, MTOT, 3, 3);
#else
            *(uint4*)&sA[1 - p][srow * 40 + sseg * 16] =
                *(const uint4*)(aRow + k0 + 32);
            *(uint4*)&sA[1 - p][srow * 40 + sseg * 16 + 8] =
                *(const uint4*)(aRow + k0 + 40);
#endif
            __builtin_prefetch(wBase[0] + k0 + 32, 0, 3);
            __builtin_prefetch(wBase[1] + k0 + 32, 0, 3);
        }

        FragAB bf[2];
#pragma unroll
        for (int sn = 0; sn < 2; sn++) {
            bf[sn].q[0] = *(const uint4*)(wBase[sn] + k0);
            bf[sn].q[1] = *(const uint4*)(wBase[sn] + k0 + 16);
        }
#pragma unroll
        for (int sm = 0; sm < 4; sm++) {
            FragAB af;
            const unsigned short* ap =
                &sA[p][(waveM * 64 + sm * 16 + l16) * 40 + lhalf * 8];
            af.q[0] = *(const uint4*)ap;
            af.q[1] = *(const uint4*)(ap + 16);
#pragma unroll
            for (int sn = 0; sn < 2; sn++)
                c[sm][sn].v = __builtin_amdgcn_wmma_f32_16x16x32_bf16(
                    false, af.v, false, bf[sn].v, (short)0, c[sm][sn].v, false, false);
        }

#if USE_TDM
        tdm_wait();
#endif
        __syncthreads();
    }

#pragma unroll
    for (int sm = 0; sm < 4; sm++) {
        const int m0 = tile_m + waveM * 64 + sm * 16;
#pragma unroll
        for (int sn = 0; sn < 2; sn++) {
            const int n = tile_n + waveN * 32 + sn * 16 + l16;
            const float bv = bias[n];
            if (mode == 0) {
                float* O = (float*)out;
#pragma unroll
                for (int r = 0; r < 8; r++) {
                    int m = m0 + r + 8 * lhalf;
                    O[(size_t)m * N + n] = c[sm][sn].f[r] + bv;
                }
            } else if (mode == 1) {
                unsigned short* O = (unsigned short*)out;
                int h = n >> 6, hd = n & 63;
#pragma unroll
                for (int r = 0; r < 8; r++) {
                    int m = m0 + r + 8 * lhalf;
                    int b = m >> 11, s = m & (SS - 1);
                    O[((size_t)(b * HH + h) * SS + s) * HDIM + hd] =
                        f32_bf16(c[sm][sn].f[r] + bv);
                }
            } else {
                unsigned short* O = (unsigned short*)out;
                int h = n >> 6, hd = n & 63;
                int mbase = m0 + 8 * lhalf;
                int b = mbase >> 11, s = mbase & (SS - 1);
                union { unsigned short us[8]; uint4 q; } pk;
#pragma unroll
                for (int r = 0; r < 8; r++) pk.us[r] = f32_bf16(c[sm][sn].f[r] + bv);
                *(uint4*)&O[((size_t)(b * HH + h) * HDIM + hd) * SS + s] = pk.q;
            }
        }
    }
}

// ---------------------------------------------------------------------------
// Causal flash attention per (b,h).
// Block: 128 threads (4 waves); each wave owns 16 query rows (q-tile = 64).
// Key blocks of 64 staged in double-buffered LDS (TDM for block kb+1 issued
// before computing block kb -> DMA hidden under scores/softmax/PV);
// online softmax; bf16 WMMA.
// Q,K: [B,H,S,HD] bf16; Vt: [B,H,HD,S] bf16; Out: [B,S,H,HD] bf16.
// ---------------------------------------------------------------------------
__global__ __launch_bounds__(128, 1) void flash_attn_kernel(
    const unsigned short* __restrict__ Q, const unsigned short* __restrict__ Kh,
    const unsigned short* __restrict__ Vt, unsigned short* __restrict__ Out) {
    __shared__ unsigned short sK[2][64 * 72];  // [key][d]
    __shared__ unsigned short sV[2][64 * 72];  // [hd][key]  (V^T)
    __shared__ unsigned short sP[4 * 16 * 72]; // per-wave P scratch [row][key]

    const int bh = blockIdx.y;  // b*H + h
    const int qt = blockIdx.x;  // q tile of 64
    const int tid = threadIdx.x;
    const int lane = tid & 31;
    const int wave = tid >> 5;
    const int l16 = lane & 15;
    const int lhalf = lane >> 4;
    const int q0 = qt * 64 + wave * 16;

    const unsigned short* Qb = Q + (size_t)bh * SS * HDIM;
    const unsigned short* Kb = Kh + (size_t)bh * SS * HDIM;
    const unsigned short* Vb = Vt + (size_t)bh * HDIM * SS;

    // Q A-fragments for d-steps 0..31, 32..63
    FragAB qf[2];
    {
        const unsigned short* qp = Qb + (size_t)(q0 + l16) * HDIM;
#pragma unroll
        for (int ks = 0; ks < 2; ks++) {
            qf[ks].q[0] = *(const uint4*)(qp + ks * 32 + lhalf * 8);
            qf[ks].q[1] = *(const uint4*)(qp + ks * 32 + lhalf * 8 + 16);
        }
    }

    FragC acc[4];
#pragma unroll
    for (int nt = 0; nt < 4; nt++)
#pragma unroll
        for (int r = 0; r < 8; r++) acc[nt].f[r] = 0.f;
    float rowM[8], rowL[8];
#pragma unroll
    for (int r = 0; r < 8; r++) { rowM[r] = -1e30f; rowL[r] = 0.f; }

#if !USE_TDM
    const int srow = tid >> 1;  // 0..63
    const int sseg = tid & 1;   // 0..1, 32 elems (64B) each
#endif
    const int nblocks = qt + 1; // causal: skip blocks past diagonal

    // prologue: stage key block 0 into buffer 0
#if USE_TDM
    tdm_load_2d_bf16(Kb + (size_t)(wave * 16) * HDIM, &sK[0][wave * 16 * 72],
                     64, 16, (unsigned long long)HDIM, HDIM, SS, 4, 3);
    tdm_load_2d_bf16(Vb + (size_t)(wave * 16) * SS, &sV[0][wave * 16 * 72],
                     64, 16, (unsigned long long)SS, SS, HDIM, 4, 3);
    tdm_wait();
#else
    {
#pragma unroll
        for (int t = 0; t < 4; t++)
            ((uint4*)&sK[0][srow * 72 + sseg * 32])[t] =
                ((const uint4*)(Kb + (size_t)srow * HDIM + sseg * 32))[t];
#pragma unroll
        for (int t = 0; t < 4; t++)
            ((uint4*)&sV[0][srow * 72 + sseg * 32])[t] =
                ((const uint4*)(Vb + (size_t)srow * SS + sseg * 32))[t];
    }
#endif
    __syncthreads();

    for (int kb = 0; kb < nblocks; kb++) {
        const int p = kb & 1;
        // issue next key block's DMA (overlaps all compute below)
        if (kb + 1 < nblocks) {
#if USE_TDM
            tdm_load_2d_bf16(Kb + (size_t)((kb + 1) * 64 + wave * 16) * HDIM,
                             &sK[1 - p][wave * 16 * 72],
                             64, 16, (unsigned long long)HDIM, HDIM, SS, 4, 3);
            tdm_load_2d_bf16(Vb + (size_t)(wave * 16) * SS + (kb + 1) * 64,
                             &sV[1 - p][wave * 16 * 72],
                             64, 16, (unsigned long long)SS, SS, HDIM, 4, 3);
#else
#pragma unroll
            for (int t = 0; t < 4; t++)
                ((uint4*)&sK[1 - p][srow * 72 + sseg * 32])[t] =
                    ((const uint4*)(Kb + (size_t)((kb + 1) * 64 + srow) * HDIM +
                                    sseg * 32))[t];
#pragma unroll
            for (int t = 0; t < 4; t++)
                ((uint4*)&sV[1 - p][srow * 72 + sseg * 32])[t] =
                    ((const uint4*)(Vb + (size_t)srow * SS + (kb + 1) * 64 +
                                    sseg * 32))[t];
#endif
        }

        // ---- scores: S = Q @ K^T  (4 n-subtiles of 16 keys) ----
        FragC sc[4];
#pragma unroll
        for (int ns = 0; ns < 4; ns++) {
            FragAB kf0, kf1;
            const unsigned short* kr = &sK[p][(ns * 16 + l16) * 72 + lhalf * 8];
            kf0.q[0] = *(const uint4*)kr;
            kf0.q[1] = *(const uint4*)(kr + 16);
            kf1.q[0] = *(const uint4*)(kr + 32);
            kf1.q[1] = *(const uint4*)(kr + 48);
            FragC z;
#pragma unroll
            for (int r = 0; r < 8; r++) z.f[r] = 0.f;
            FragC s_;
            s_.v = __builtin_amdgcn_wmma_f32_16x16x32_bf16(
                false, qf[0].v, false, kf0.v, (short)0, z.v, false, false);
            s_.v = __builtin_amdgcn_wmma_f32_16x16x32_bf16(
                false, qf[1].v, false, kf1.v, (short)0, s_.v, false, false);
            sc[ns] = s_;
        }

        // ---- scale + causal mask (only diagonal block can mask) ----
        const float scale = 0.125f;  // 1/sqrt(64)
        if (kb == qt) {
#pragma unroll
            for (int ns = 0; ns < 4; ns++)
#pragma unroll
                for (int r = 0; r < 8; r++) {
                    int kidx = kb * 64 + ns * 16 + l16;
                    int qidx = q0 + r + 8 * lhalf;
                    float v = sc[ns].f[r] * scale;
                    sc[ns].f[r] = (kidx > qidx) ? -1e30f : v;
                }
        } else {
#pragma unroll
            for (int ns = 0; ns < 4; ns++)
#pragma unroll
                for (int r = 0; r < 8; r++) sc[ns].f[r] *= scale;
        }

        // ---- online softmax (per-row over 64 keys; 16-lane reductions) ----
#pragma unroll
        for (int r = 0; r < 8; r++) {
            float mx = fmaxf(fmaxf(sc[0].f[r], sc[1].f[r]),
                             fmaxf(sc[2].f[r], sc[3].f[r]));
#pragma unroll
            for (int off = 1; off < 16; off <<= 1)
                mx = fmaxf(mx, __shfl_xor(mx, off, 32));
            float mnew = fmaxf(rowM[r], mx);
            float corr = __expf(rowM[r] - mnew);
            rowM[r] = mnew;
            float lsum = 0.f;
#pragma unroll
            for (int ns = 0; ns < 4; ns++) {
                float pv = __expf(sc[ns].f[r] - mnew);
                sc[ns].f[r] = pv;
                lsum += pv;
            }
#pragma unroll
            for (int off = 1; off < 16; off <<= 1)
                lsum += __shfl_xor(lsum, off, 32);
            rowL[r] = rowL[r] * corr + lsum;
#pragma unroll
            for (int nt = 0; nt < 4; nt++) acc[nt].f[r] *= corr;
        }

        // ---- C-layout -> A-layout relayout of P via per-wave LDS buffer ----
        unsigned short* pb = sP + wave * (16 * 72);
#pragma unroll
        for (int ns = 0; ns < 4; ns++)
#pragma unroll
            for (int r = 0; r < 8; r++)
                pb[(r + 8 * lhalf) * 72 + ns * 16 + l16] = f32_bf16(sc[ns].f[r]);
        // DS ops are in-order within a wave: read-after-write is safe.

        // ---- acc += P @ V  (K = 64 keys, 2 steps of 32) ----
#pragma unroll
        for (int ks = 0; ks < 2; ks++) {
            FragAB pf;
            const unsigned short* pr = pb + l16 * 72 + ks * 32 + lhalf * 8;
            pf.q[0] = *(const uint4*)pr;
            pf.q[1] = *(const uint4*)(pr + 16);
#pragma unroll
            for (int nt = 0; nt < 4; nt++) {
                FragAB vf;
                const unsigned short* vr =
                    &sV[p][(nt * 16 + l16) * 72 + ks * 32 + lhalf * 8];
                vf.q[0] = *(const uint4*)vr;
                vf.q[1] = *(const uint4*)(vr + 16);
                acc[nt].v = __builtin_amdgcn_wmma_f32_16x16x32_bf16(
                    false, pf.v, false, vf.v, (short)0, acc[nt].v, false, false);
            }
        }

#if USE_TDM
        tdm_wait();
#endif
        __syncthreads();
    }

    // ---- normalize and write [B,S,H,HD] bf16 ----
    const int b = bh >> 4;
    const int h = bh & 15;
#pragma unroll
    for (int nt = 0; nt < 4; nt++) {
        int hd = nt * 16 + l16;
#pragma unroll
        for (int r = 0; r < 8; r++) {
            int s = q0 + r + 8 * lhalf;
            float o = acc[nt].f[r] * (1.0f / rowL[r]);
            Out[((size_t)(b * SS + s) * HH + h) * HDIM + hd] = f32_bf16(o);
        }
    }
}

// ---------------------------------------------------------------------------
extern "C" void kernel_launch(void* const* d_in, const int* in_sizes, int n_in,
                              void* d_out, int out_size, void* d_ws, size_t ws_size,
                              hipStream_t stream) {
    const float* q = (const float*)d_in[0];
    const float* k = (const float*)d_in[1];
    const float* v = (const float*)d_in[2];
    // d_in[3]: causal mask — handled analytically
    const float* w_q = (const float*)d_in[4];
    const float* b_q = (const float*)d_in[5];
    const float* w_k = (const float*)d_in[6];
    const float* b_k = (const float*)d_in[7];
    const float* w_v = (const float*)d_in[8];
    const float* b_v = (const float*)d_in[9];
    const float* w_o = (const float*)d_in[10];
    const float* b_o = (const float*)d_in[11];
    float* out = (float*)d_out;

    char* ws = (char*)d_ws;
    const size_t MiB = 1ull << 20;
    unsigned short* xq = (unsigned short*)(ws + 0 * MiB);    // [M,K] bf16 (16 MiB)
    unsigned short* xk = (unsigned short*)(ws + 16 * MiB);
    unsigned short* xv = (unsigned short*)(ws + 32 * MiB);
    unsigned short* wqb = (unsigned short*)(ws + 48 * MiB);  // [N,K] bf16 (2 MiB)
    unsigned short* wkb = (unsigned short*)(ws + 50 * MiB);
    unsigned short* wvb = (unsigned short*)(ws + 52 * MiB);
    unsigned short* wob = (unsigned short*)(ws + 54 * MiB);
    unsigned short* qh = (unsigned short*)(ws + 56 * MiB);   // [B,H,S,HD] (16 MiB)
    unsigned short* kh = (unsigned short*)(ws + 72 * MiB);   // [B,H,S,HD]
    unsigned short* vt = (unsigned short*)(ws + 88 * MiB);   // [B,H,HD,S]
    unsigned short* ao = (unsigned short*)(ws + 104 * MiB);  // [B,S,H*HD] = [M,D]

    // 1) convert activations + weights to bf16
    const int nAct8 = MTOT * DD / 8;  // 1,048,576
    const int nW8 = DD * DD / 8;      // 131,072
    cvt_f32_bf16_kernel<<<nAct8 / 256, 256, 0, stream>>>(q, xq, nAct8);
    cvt_f32_bf16_kernel<<<nAct8 / 256, 256, 0, stream>>>(k, xk, nAct8);
    cvt_f32_bf16_kernel<<<nAct8 / 256, 256, 0, stream>>>(v, xv, nAct8);
    cvt_f32_bf16_kernel<<<nW8 / 256, 256, 0, stream>>>(w_q, wqb, nW8);
    cvt_f32_bf16_kernel<<<nW8 / 256, 256, 0, stream>>>(w_k, wkb, nW8);
    cvt_f32_bf16_kernel<<<nW8 / 256, 256, 0, stream>>>(w_v, wvb, nW8);
    cvt_f32_bf16_kernel<<<nW8 / 256, 256, 0, stream>>>(w_o, wob, nW8);

    // 2) QKV projections (outputs in head layouts; V transposed)
    dim3 ggrid(MTOT / 128, DD / 128);  // (64, 8)
    gemm_bf16_kernel<<<ggrid, 256, 0, stream>>>(xq, wqb, b_q, qh, 1);
    gemm_bf16_kernel<<<ggrid, 256, 0, stream>>>(xk, wkb, b_k, kh, 1);
    gemm_bf16_kernel<<<ggrid, 256, 0, stream>>>(xv, wvb, b_v, vt, 2);

    // 3) causal flash attention
    dim3 agrid(SS / 64, BB * HH);  // (32, 64)
    flash_attn_kernel<<<agrid, 128, 0, stream>>>(qh, kh, vt, ao);

    // 4) output projection -> f32 d_out
    gemm_bf16_kernel<<<ggrid, 256, 0, stream>>>(ao, wob, b_o, out, 0);
}
